// nn_model_52407190945848
// MI455X (gfx1250) — matrix operand order repaired and checked by reading, not compiled, not run on hardware
//
#include <hip/hip_runtime.h>
#include <math.h>

typedef __attribute__((ext_vector_type(16))) _Float16 v16h;
typedef __attribute__((ext_vector_type(8)))  _Float16 v8h;
typedef __attribute__((ext_vector_type(8)))  float    v8f;
typedef __attribute__((ext_vector_type(4)))  float    v4f;

namespace {
constexpr int TSTEPS   = 256;   // timestamp
constexpr int UNITS    = 50;
constexpr int GATES    = 200;   // 4*UNITS, columns permuted as n' = 4*u + gate
constexpr int NPAD     = 256;   // 16 WMMA N-tiles: every wave owns exactly 4 -> no guards
constexpr int K0PAD    = 64;    // layer0 K: h0(50) zero-padded
constexpr int K1PAD    = 128;   // layer1 K: [nh0(50)|pad|h1(50)|pad]
constexpr int ROWS     = 16;    // batch rows per workgroup (one WMMA M-tile)
constexpr int NTHREADS = 128;   // 4 wave32
constexpr int TPW      = 4;     // N-tiles per wave (uniform)

// LDS layout (bytes, all 16B aligned)
constexpr int OFF_WT0 = 0;                              // U0^T  [NPAD][K0PAD] f16 (cols permuted)
constexpr int OFF_WT1 = OFF_WT0 + NPAD*K0PAD*2;         // [W1;U1]^T [NPAD][K1PAD] f16 (cols permuted)
constexpr int OFF_A0  = OFF_WT1 + NPAD*K1PAD*2;         // h0 state as f16 A-operand [16][64]
constexpr int OFF_A1  = OFF_A0  + ROWS*K0PAD*2;         // [nh0|h1] f16 A-operand [16][128]
constexpr int OFF_ZB  = OFF_A1  + ROWS*K1PAD*2;         // z scratch f32, n-major [NPAD][16]
constexpr int OFF_C0  = OFF_ZB  + NPAD*ROWS*4;          // c0 f32 [u*16+m]
constexpr int OFF_C1  = OFF_C0  + ROWS*UNITS*4;         // c1 f32 [u*16+m]
constexpr int OFF_W0S = OFF_C1  + ROWS*UNITS*4;         // W0 row f32 [256] (permuted)
constexpr int OFF_B0S = OFF_W0S + NPAD*4;               // b0 f32 [256] (permuted)
constexpr int OFF_B1S = OFF_B0S + NPAD*4;               // b1 f32 [256] (permuted)
constexpr int OFF_XT  = OFF_B1S + NPAD*4;               // x tile f32 [16][256]
constexpr int SMEM_BYTES = OFF_XT + ROWS*TSTEPS*4;      // ~147 KB of 320 KB/WGP (2 WGs/WGP)
}

__device__ __forceinline__ float sigf(float z)     { return 1.0f / (1.0f + __expf(-z)); }
__device__ __forceinline__ float tanhfast(float z) { return 1.0f - 2.0f / (__expf(2.0f*z) + 1.0f); }

__device__ __forceinline__ v16h cat8(v8h lo, v8h hi) {
    return __builtin_shufflevector(lo, hi, 0,1,2,3,4,5,6,7,8,9,10,11,12,13,14,15);
}

__global__ __launch_bounds__(NTHREADS, 1)
void lstm2_wmma_persistent(const float* __restrict__ x,
                           const float* __restrict__ W0, const float* __restrict__ U0,
                           const float* __restrict__ b0,
                           const float* __restrict__ W1, const float* __restrict__ U1,
                           const float* __restrict__ b1,
                           const float* __restrict__ Wd, const float* __restrict__ bd,
                           float* __restrict__ out)
{
    extern __shared__ char smem[];
    _Float16* wt0 = (_Float16*)(smem + OFF_WT0);
    _Float16* wt1 = (_Float16*)(smem + OFF_WT1);
    _Float16* a0h = (_Float16*)(smem + OFF_A0);
    _Float16* a1h = (_Float16*)(smem + OFF_A1);
    float*    zb  = (float*)(smem + OFF_ZB);
    float*    c0s = (float*)(smem + OFF_C0);
    float*    c1s = (float*)(smem + OFF_C1);
    float*    w0s = (float*)(smem + OFF_W0S);
    float*    b0s = (float*)(smem + OFF_B0S);
    float*    b1s = (float*)(smem + OFF_B1S);
    float*    xt  = (float*)(smem + OFF_XT);

    const int tid  = threadIdx.x;
    const int row0 = blockIdx.x * ROWS;

    // ---- stage weights once: transposed [N'][K] f16, gate-interleaved columns n' = 4u+g ----
    for (int i = tid; i < NPAD*K0PAD; i += NTHREADS) {
        int np = i >> 6, k = i & 63;            // permuted col, K
        int u = np >> 2, g = np & 3;
        float v = (np < GATES && k < UNITS) ? U0[k*GATES + g*UNITS + u] : 0.0f;
        wt0[i] = (_Float16)v;
    }
    for (int i = tid; i < NPAD*K1PAD; i += NTHREADS) {
        int np = i >> 7, k = i & 127;
        int u = np >> 2, g = np & 3;
        float v = 0.0f;
        if (np < GATES) {
            if (k < UNITS)                            v = W1[k*GATES + g*UNITS + u];
            else if (k >= K0PAD && k < K0PAD + UNITS) v = U1[(k - K0PAD)*GATES + g*UNITS + u];
        }
        wt1[i] = (_Float16)v;
    }
    for (int i = tid; i < NPAD; i += NTHREADS) {
        int u = i >> 2, g = i & 3;
        bool inr = (i < GATES);
        int ng = g*UNITS + u;                    // original column index
        w0s[i] = inr ? W0[ng] : 0.0f;
        b0s[i] = inr ? b0[ng] : 0.0f;
        b1s[i] = inr ? b1[ng] : 0.0f;
    }
    // x tile for all timesteps + zeroed recurrent state
    for (int i = tid; i < ROWS*TSTEPS; i += NTHREADS) {
        int m = i >> 8, tt = i & 255;
        xt[i] = x[(row0 + m)*TSTEPS + tt];
    }
    for (int i = tid; i < ROWS*K0PAD; i += NTHREADS) a0h[i] = (_Float16)0.0f;
    for (int i = tid; i < ROWS*K1PAD; i += NTHREADS) a1h[i] = (_Float16)0.0f;
    for (int i = tid; i < ROWS*UNITS; i += NTHREADS) { c0s[i] = 0.0f; c1s[i] = 0.0f; }
    __syncthreads();

    const int lane  = tid & 31;
    const int wave  = tid >> 5;
    const int nlo   = lane & 15;    // A row / B column / C column within tile
    const int half  = lane >> 4;    // lane-group selector
    const int kselA = half * 16;    // A frag: 16 contiguous K halves of row nlo
    const int kselB = half * 16;    // B frag: 16 contiguous K halves of column nlo

    // ---- hoist ALL time-invariant operands into registers (live for 256 steps) ----
    float bias0r[TPW], bias1r[TPW], w0r[TPW];
    v16h b0f[K0PAD/32][TPW];        // layer-0 weight fragments:  2*4*8 = 64 VGPRs
    v16h b1f[K1PAD/32][TPW];        // layer-1 weight fragments:  4*4*8 = 128 VGPRs
#pragma unroll
    for (int j = 0; j < TPW; ++j) {
        const int ncol = (wave + 4*j)*16 + nlo;
        bias0r[j] = b0s[ncol];
        bias1r[j] = b1s[ncol];
        w0r[j]    = w0s[ncol];
#pragma unroll
        for (int kk = 0; kk < K0PAD/32; ++kk) {
            const _Float16* bp = &wt0[ncol*K0PAD + kk*32 + kselB];
            b0f[kk][j] = cat8(*(const v8h*)bp, *(const v8h*)(bp + 8));
        }
#pragma unroll
        for (int kk = 0; kk < K1PAD/32; ++kk) {
            const _Float16* bp = &wt1[ncol*K1PAD + kk*32 + kselB];
            b1f[kk][j] = cat8(*(const v8h*)bp, *(const v8h*)(bp + 8));
        }
    }

    for (int t = 0; t < TSTEPS; ++t) {
        // per-lane x_t values for its 8 C rows, shared across all N-tiles
        float xr[8];
#pragma unroll
        for (int r = 0; r < 8; ++r) xr[r] = xt[(r + half*8)*TSTEPS + t];

        // ---------------- layer 0: z = h0 @ U0 + x_t (x) W0 + b0 ----------------
        {
            v8f acc[TPW];
#pragma unroll
            for (int j = 0; j < TPW; ++j)
#pragma unroll
                for (int r = 0; r < 8; ++r)
                    acc[j][r] = bias0r[j] + xr[r] * w0r[j];

#pragma unroll
            for (int kk = 0; kk < K0PAD/32; ++kk) {
                const _Float16* ap = &a0h[nlo*K0PAD + kk*32 + kselA];
                v16h af = cat8(*(const v8h*)ap, *(const v8h*)(ap + 8));
#pragma unroll
                for (int j = 0; j < TPW; ++j)     // 4 independent chains hide D->C RAW latency
                    acc[j] = __builtin_amdgcn_wmma_f32_16x16x32_f16(
                                 false, af, false, b0f[kk][j], (short)0, acc[j], false, false);
            }
#pragma unroll
            for (int j = 0; j < TPW; ++j) {
                float* zp = &zb[((wave + 4*j)*16 + nlo)*ROWS + half*8];   // 32B contiguous
                *(v4f*)zp       = v4f{acc[j][0], acc[j][1], acc[j][2], acc[j][3]};
                *(v4f*)(zp + 4) = v4f{acc[j][4], acc[j][5], acc[j][6], acc[j][7]};
            }
        }
        __syncthreads();

        // gates layer 0: columns interleaved (i,f,g,o) at n' = 4u..4u+3, shift/mask indexing
        for (int idx = tid; idx < ROWS*UNITS; idx += NTHREADS) {
            int u = idx >> 4, m = idx & 15;
            float zi = zb[(4*u + 0)*ROWS + m];
            float zf = zb[(4*u + 1)*ROWS + m];
            float zg = zb[(4*u + 2)*ROWS + m];
            float zo = zb[(4*u + 3)*ROWS + m];
            float cn = sigf(zf)*c0s[idx] + sigf(zi)*tanhfast(zg);
            float hn = sigf(zo)*tanhfast(cn);
            a1h[m*K1PAD + u] = (_Float16)hn;          // layer-1 input: UN-masked nh0
            if (xt[m*TSTEPS + t] != -1.0f) {          // mask: freeze state where x==MASK_VAL
                c0s[idx] = cn;
                a0h[m*K0PAD + u] = (_Float16)hn;
            }
        }
        __syncthreads();

        // ---------------- layer 1: z = [nh0 | h1] @ [W1 ; U1] + b1 ----------------
        {
            v8f acc[TPW];
#pragma unroll
            for (int j = 0; j < TPW; ++j)
#pragma unroll
                for (int r = 0; r < 8; ++r) acc[j][r] = bias1r[j];

#pragma unroll
            for (int kk = 0; kk < K1PAD/32; ++kk) {
                const _Float16* ap = &a1h[nlo*K1PAD + kk*32 + kselA];
                v16h af = cat8(*(const v8h*)ap, *(const v8h*)(ap + 8));
#pragma unroll
                for (int j = 0; j < TPW; ++j)
                    acc[j] = __builtin_amdgcn_wmma_f32_16x16x32_f16(
                                 false, af, false, b1f[kk][j], (short)0, acc[j], false, false);
            }
#pragma unroll
            for (int j = 0; j < TPW; ++j) {
                float* zp = &zb[((wave + 4*j)*16 + nlo)*ROWS + half*8];
                *(v4f*)zp       = v4f{acc[j][0], acc[j][1], acc[j][2], acc[j][3]};
                *(v4f*)(zp + 4) = v4f{acc[j][4], acc[j][5], acc[j][6], acc[j][7]};
            }
        }
        __syncthreads();

        // gates layer 1
        for (int idx = tid; idx < ROWS*UNITS; idx += NTHREADS) {
            int u = idx >> 4, m = idx & 15;
            float zi = zb[(4*u + 0)*ROWS + m];
            float zf = zb[(4*u + 1)*ROWS + m];
            float zg = zb[(4*u + 2)*ROWS + m];
            float zo = zb[(4*u + 3)*ROWS + m];
            float cn = sigf(zf)*c1s[idx] + sigf(zi)*tanhfast(zg);
            float hn = sigf(zo)*tanhfast(cn);
            if (xt[m*TSTEPS + t] != -1.0f) {
                c1s[idx] = cn;
                a1h[m*K1PAD + K0PAD + u] = (_Float16)hn;   // h1 state
            }
        }
        __syncthreads();
    }

    // dense 50->1 then softmax over a single class (== 1.0 for every row)
    if (tid < ROWS) {
        float acc = bd[0];
        for (int u = 0; u < UNITS; ++u)
            acc += (float)a1h[tid*K1PAD + K0PAD + u] * Wd[u];
        float mx = acc;                  // row max (single element)
        float e  = __expf(acc - mx);     // exp(0) = 1
        out[row0 + tid] = e / e;         // normalize over the single column
    }
}

extern "C" void kernel_launch(void* const* d_in, const int* in_sizes, int n_in,
                              void* d_out, int out_size, void* d_ws, size_t ws_size,
                              hipStream_t stream) {
    (void)n_in; (void)out_size; (void)d_ws; (void)ws_size;
    const float* x  = (const float*)d_in[0];
    // d_in[1] = timestamp (== 256, baked in as TSTEPS)
    const float* W0 = (const float*)d_in[2];
    const float* U0 = (const float*)d_in[3];
    const float* b0 = (const float*)d_in[4];
    const float* W1 = (const float*)d_in[5];
    const float* U1 = (const float*)d_in[6];
    const float* b1 = (const float*)d_in[7];
    const float* Wd = (const float*)d_in[8];
    const float* bd = (const float*)d_in[9];

    const int Bn = in_sizes[0] / TSTEPS;       // 4096
    dim3 grid(Bn / ROWS), block(NTHREADS);
    lstm2_wmma_persistent<<<grid, block, SMEM_BYTES, stream>>>(
        x, W0, U0, b0, W1, U1, b1, Wd, bd, (float*)d_out);
}
